// RNNAttModel_80032420594136
// MI455X (gfx1250) — matrix operand order repaired and checked by reading, not compiled, not run on hardware
//
#include <hip/hip_runtime.h>
#include <math.h>
#include <stdint.h>

typedef _Float16 half_t;
typedef __attribute__((ext_vector_type(16))) _Float16 v16h;
typedef __attribute__((ext_vector_type(8)))  float    v8f;
typedef __attribute__((ext_vector_type(4)))  unsigned int v4u;

// ---------------- constants ----------------
#define BSZ   1024
#define FEATN 66
#define DMN   512
#define HSN   50
// derived: VL=35, VN=16, rows = BSZ*FEATN = 67584

__device__ __forceinline__ float dct_w(int d, int l) {
  // orthonormal DCT-II, N=35; inverse == transpose
  float wd = (d == 0) ? 0.16903085094570331f : 0.23904572186687872f; // sqrt(1/35), sqrt(2/35)
  return wd * cosf(3.14159265358979323846f * ((float)l + 0.5f) * (float)d * (1.0f / 35.0f));
}
__device__ __forceinline__ float sigmoidf(float x) { return 1.0f / (1.0f + expf(-x)); }

// async global->LDS 16B copy (CDNA5: GLOBAL_LOAD_ASYNC_TO_LDS_B128, ASYNCcnt-tracked)
__device__ __forceinline__ void async_copy16(const void* gptr, void* lptr) {
  unsigned       loff = (unsigned)(uintptr_t)lptr;          // generic LDS addr low 32 = LDS offset
  unsigned long long g = (unsigned long long)(uintptr_t)gptr;
  asm volatile("global_load_async_to_lds_b128 %0, %1, off" :: "v"(loff), "v"(g) : "memory");
}
__device__ __forceinline__ void wait_async0() {
  asm volatile("s_wait_asynccnt 0" ::: "memory");
}

// ---------------- LSTM (single step, h0=c0=0, forget gate unused) ----------------
__global__ void lstm_kernel(const float* __restrict__ src, const float* __restrict__ Wih,
                            const float* __restrict__ bih, const float* __restrict__ bhh,
                            float* __restrict__ hOut, float* __restrict__ cOut, int total) {
  int idx = blockIdx.x * blockDim.x + threadIdx.x;
  if (idx >= total) return;
  int r = idx / HSN, u = idx % HSN;
  const float* x  = src + (size_t)r * 10;
  const float* wi = Wih + (size_t)u * 10;
  const float* wg = Wih + (size_t)(100 + u) * 10;
  const float* wo = Wih + (size_t)(150 + u) * 10;
  float gi = bih[u] + bhh[u];
  float gg = bih[100 + u] + bhh[100 + u];
  float go = bih[150 + u] + bhh[150 + u];
  for (int i = 0; i < 10; ++i) {
    float xv = x[i];
    gi += xv * wi[i]; gg += xv * wg[i]; go += xv * wo[i];
  }
  float c = sigmoidf(gi) * tanhf(gg);
  float h = sigmoidf(go) * tanhf(c);
  hOut[idx] = h; cOut[idx] = c;
}

// ---------------- f32 -> f16 weight / matrix packs ----------------
__global__ void pack_w_kernel(const float* __restrict__ src, half_t* __restrict__ dst,
                              int srcR, int srcC, int dstR, int dstC, int transpose, int total) {
  int idx = blockIdx.x * blockDim.x + threadIdx.x;
  if (idx >= total) return;
  int r = idx / dstC, c = idx % dstC;
  float v = 0.0f;
  if (transpose) { if (r < srcC && c < srcR) v = src[(size_t)c * srcC + r]; }
  else           { if (r < srcR && c < srcC) v = src[(size_t)r * srcC + c]; }
  dst[(size_t)r * dstC + c] = (half_t)v;
}

__global__ void cvt_f32_f16_kernel(const float* __restrict__ in, half_t* __restrict__ out, int n) {
  int i = blockIdx.x * blockDim.x + threadIdx.x;
  if (i < n) out[i] = (half_t)in[i];
}

// im2col for conv1 (input = h viewed as src_T[b,f,t]); rows m=(b,p), K=66*6 padded to 416
__global__ void pack_conv1_kernel(const float* __restrict__ h, half_t* __restrict__ A,
                                  int P, int tbase, int total) {
  int idx = blockIdx.x * blockDim.x + threadIdx.x;
  if (idx >= total) return;
  int k = idx % 416, m = idx / 416;
  int b = m / P, p = m % P;
  float v = 0.0f;
  if (k < 396) {
    int f = k / 6, j = k % 6;
    v = h[(size_t)b * 3300 + (size_t)(tbase + p + j) * 66 + f] * 1e-3f;
  }
  A[(size_t)m * 416 + k] = (half_t)v;
}

// im2col for conv2: input O1 laid out (b, P, 512); rows m=(b,p2), K = 512*5 = 2560
__global__ void pack_conv2_kernel(const float* __restrict__ O1, half_t* __restrict__ A,
                                  int P, int P2, int total) {
  int idx = blockIdx.x * blockDim.x + threadIdx.x;
  if (idx >= total) return;
  int k = idx % 2560, m = idx / 2560;
  int b = m / P2, p2 = m % P2;
  int c = k / 5, j = k % 5;
  A[(size_t)m * 2560 + k] = (half_t)O1[((size_t)b * P + p2 + j) * 512 + c];
}

// ---------------- generic WMMA GEMM: C(f32, MxN) = A(f16 MxK) @ B(f16 KxN), opt. relu ----------------
// BM=BN=64, BK=32, 128 threads (4 waves, 2x2 wave grid, each wave 2x2 16x16 tiles).
// Double-buffered LDS fed by async global->LDS copies (ASYNCcnt), one barrier per k-step.
__global__ __launch_bounds__(128)
void gemm_wmma_kernel(const half_t* __restrict__ A, const half_t* __restrict__ B,
                      float* __restrict__ C, int M, int N, int K, int relu) {
  __shared__ __attribute__((aligned(16))) half_t As[2][64 * 32];
  __shared__ __attribute__((aligned(16))) half_t Bs[2][32 * 64];
  (void)M;
  const int t = threadIdx.x;
  const int wv = t >> 5, lane = t & 31;
  const int m0 = blockIdx.y * 64, n0 = blockIdx.x * 64;
  const int wm = (wv >> 1) * 32, wn = (wv & 1) * 32;
  // per-thread 16B chunk coords (uniform per wave; 2 chunks per operand per tile)
  const int ar0 = t >> 2, ac = (t & 3) << 3;   // A rows ar0, ar0+32; cols ac..ac+7
  const int br0 = t >> 3, bc = (t & 7) << 3;   // B rows br0, br0+16; cols bc..bc+7
  const int nk = K >> 5;

  const v8f zero = {0.f, 0.f, 0.f, 0.f, 0.f, 0.f, 0.f, 0.f};
  v8f acc[2][2];
  acc[0][0] = zero; acc[0][1] = zero; acc[1][0] = zero; acc[1][1] = zero;
  const int arow = lane & 15, kb = (lane >> 4) << 3;

  // prologue: async-stage tile 0 into buffer 0
  {
    const int k0 = 0;
    async_copy16(&A[(size_t)(m0 + ar0) * K + k0 + ac],      &As[0][ar0 * 32 + ac]);
    async_copy16(&A[(size_t)(m0 + ar0 + 32) * K + k0 + ac], &As[0][(ar0 + 32) * 32 + ac]);
    async_copy16(&B[(size_t)(k0 + br0) * N + n0 + bc],      &Bs[0][br0 * 64 + bc]);
    async_copy16(&B[(size_t)(k0 + br0 + 16) * N + n0 + bc], &Bs[0][(br0 + 16) * 64 + bc]);
  }

  for (int kt = 0; kt < nk; ++kt) {
    const int buf = kt & 1;
    wait_async0();          // my async writes for tile kt have landed
    __syncthreads();        // everyone's have; also: all reads of buf^1 from kt-1 are done
    if (kt + 1 < nk) {      // async-stage next tile into the other buffer
      const int k0 = (kt + 1) << 5;
      async_copy16(&A[(size_t)(m0 + ar0) * K + k0 + ac],      &As[buf ^ 1][ar0 * 32 + ac]);
      async_copy16(&A[(size_t)(m0 + ar0 + 32) * K + k0 + ac], &As[buf ^ 1][(ar0 + 32) * 32 + ac]);
      async_copy16(&B[(size_t)(k0 + br0) * N + n0 + bc],      &Bs[buf ^ 1][br0 * 64 + bc]);
      async_copy16(&B[(size_t)(k0 + br0 + 16) * N + n0 + bc], &Bs[buf ^ 1][(br0 + 16) * 64 + bc]);
    }
    v16h af[2], bf[2];
    for (int ti = 0; ti < 2; ++ti) {                     // A frags: rows wm+ti*16..+15
      const half_t* ap = &As[buf][(wm + ti * 16 + arow) * 32];
      ((v4u*)&af[ti])[0] = *(const v4u*)&ap[kb];
      ((v4u*)&af[ti])[1] = *(const v4u*)&ap[kb + 16];
    }
    for (int tj = 0; tj < 2; ++tj) {                     // B frags: lane holds N=lane%16 column, K in table order
      const half_t* bp = &Bs[buf][wn + tj * 16 + (lane & 15)];
      for (int s = 0; s < 16; ++s)
        bf[tj][s] = bp[(kb + (s & 7) + ((s >> 3) << 4)) * 64];
    }
    for (int ti = 0; ti < 2; ++ti)
      for (int tj = 0; tj < 2; ++tj)
        acc[ti][tj] = __builtin_amdgcn_wmma_f32_16x16x32_f16(
            false, af[ti], false, bf[tj], (short)0, acc[ti][tj], false, false);
  }
  const int crow = (lane >> 4) << 3;   // M = v + 8*(lane/16)
  const int ccol = lane & 15;
  for (int ti = 0; ti < 2; ++ti)
    for (int tj = 0; tj < 2; ++tj)
      for (int v = 0; v < 8; ++v) {
        float x = acc[ti][tj][v];
        if (relu) x = fmaxf(x, 0.0f);
        C[(size_t)(m0 + wm + ti * 16 + crow + v) * N + (n0 + wn + tj * 16 + ccol)] = x;
      }
}

// ---------------- batched gconv node-mix: out[b,n,f] = epilogue( sum_m att[n,m]*XW[b,m,f] ) ----------------
// att pre-packed f16 (80x96, zero-padded) staged via async global->LDS.
// XW f32 tile converted to f16 in LDS. emode 0: tanh(bn(acc+gb)); 1: +res; 2: acc+gb+res (gc7)
__global__ __launch_bounds__(128)
void att_wmma_kernel(const half_t* __restrict__ attP,
                     const float* __restrict__ XW, int ldsrc,
                     const float* __restrict__ gb,
                     const float* __restrict__ bng, const float* __restrict__ bnb,
                     const float* __restrict__ res, int ldres,
                     float* __restrict__ out, int ldout, int colMax, int emode) {
  __shared__ __attribute__((aligned(16))) half_t attS[80 * 96];
  __shared__ __attribute__((aligned(16))) half_t xwS[96 * 64];
  const int t = threadIdx.x;
  const int b = blockIdx.y;
  const int n0 = blockIdx.x * 64;
  const float RSQ = 0.9999950000374997f; // 1/sqrt(1+1e-5)

  // att slice: contiguous 7680 halves = 960 x 16B, async copy (uniform trip count per wave)
  for (int i = t; i < 960; i += 128)
    async_copy16(attP + (size_t)i * 8, &attS[i * 8]);
  // xw tile 96x64: f32 -> f16 conversion in flight, pad rows >= 66
  for (int i = t; i < 1536; i += 128) {
    int r = i >> 4, c = (i & 15) << 2;
    float4 v = make_float4(0.f, 0.f, 0.f, 0.f);
    if (r < 66) v = *(const float4*)&XW[((size_t)b * 66 + r) * ldsrc + n0 + c];
    xwS[r * 64 + c + 0] = (half_t)v.x;
    xwS[r * 64 + c + 1] = (half_t)v.y;
    xwS[r * 64 + c + 2] = (half_t)v.z;
    xwS[r * 64 + c + 3] = (half_t)v.w;
  }
  wait_async0();
  __syncthreads();

  const int wv = t >> 5, lane = t & 31;
  const int arow = lane & 15, kb = (lane >> 4) << 3;
  const v8f zero = {0.f, 0.f, 0.f, 0.f, 0.f, 0.f, 0.f, 0.f};
  for (int tile = wv; tile < 20; tile += 4) {   // 5 M-tiles x 4 col-tiles
    int mt = tile >> 2, ct = tile & 3;
    v8f acc = zero;
    for (int k0 = 0; k0 < 96; k0 += 32) {
      v16h a, bf;
      const half_t* ap = &attS[(mt * 16 + arow) * 96 + k0];
      ((v4u*)&a)[0] = *(const v4u*)&ap[kb];
      ((v4u*)&a)[1] = *(const v4u*)&ap[kb + 16];
      const half_t* bp = &xwS[k0 * 64 + ct * 16 + (lane & 15)];
      for (int s = 0; s < 16; ++s)
        bf[s] = bp[(kb + (s & 7) + ((s >> 3) << 4)) * 64];
      acc = __builtin_amdgcn_wmma_f32_16x16x32_f16(false, a, false, bf, (short)0, acc, false, false);
    }
    int fcol = n0 + ct * 16 + (lane & 15);
    int nbase = mt * 16 + ((lane >> 4) << 3);
    for (int v = 0; v < 8; ++v) {
      int n = nbase + v;
      if (n < 66 && fcol < colMax) {
        size_t row = (size_t)b * 66 + n;
        float x = acc[v];
        if (emode == 2) {
          x = x + gb[fcol] + res[row * ldres + fcol];
        } else {
          x = tanhf((x + gb[fcol]) * bng[n * 512 + fcol] * RSQ + bnb[n * 512 + fcol]);
          if (emode == 1) x += res[row * ldres + fcol];
        }
        out[row * ldout + fcol] = x;
      }
    }
  }
}

// ---------------- DCT of value windows: V[b,v,f*10+d] = sum_l dct[d,l]*h[b,v+l,f] ----------------
__global__ void value_dct_kernel(const float* __restrict__ h, float* __restrict__ V, int total) {
  __shared__ float T[350];
  for (int i = threadIdx.x; i < 350; i += blockDim.x) T[i] = dct_w(i / 35, i % 35);
  __syncthreads();
  int idx = blockIdx.x * blockDim.x + threadIdx.x;
  if (idx >= total) return;
  int d = idx % 10, f = (idx / 10) % 66, v = (idx / 660) % 16, b = idx / 10560;
  const float* hp = h + (size_t)b * 3300 + f;
  float s = 0.f;
  for (int l = 0; l < 35; ++l) s += T[d * 35 + l] * hp[(size_t)(v + l) * 66];
  V[idx] = s;
}

// ---------------- attention score + normalize: attW[b,k] ----------------
__global__ void score_kernel(const float* __restrict__ Q, const float* __restrict__ Kt,
                             float* __restrict__ attW) {
  __shared__ float sS[17];
  int b = blockIdx.x;
  int lane = threadIdx.x & 31, k = threadIdx.x >> 5;  // 16 waves
  const float* q  = Q  + (size_t)b * 512;
  const float* kt = Kt + ((size_t)b * 16 + k) * 512;
  float p = 0.f;
  for (int d = lane; d < 512; d += 32) p += q[d] * kt[d];
  for (int o = 16; o > 0; o >>= 1) p += __shfl_xor(p, o, 32);
  if (lane == 0) sS[k] = p + 1e-15f;
  __syncthreads();
  if (threadIdx.x == 0) {
    float s = 0.f;
    for (int i = 0; i < 16; ++i) s += sS[i];
    sS[16] = s;
  }
  __syncthreads();
  if (threadIdx.x < 16) attW[(size_t)b * 16 + threadIdx.x] = sS[threadIdx.x] / sS[16];
}

// ---------------- dct_in: gcnin[b*66+f, 0..9]; PAD_IDX = [40..49, 49 x25] ----------------
__global__ void dct_in_kernel(const float* __restrict__ h, float* __restrict__ gcnin, int total) {
  __shared__ float T[350];
  for (int i = threadIdx.x; i < 350; i += blockDim.x) T[i] = dct_w(i / 35, i % 35);
  __syncthreads();
  int idx = blockIdx.x * blockDim.x + threadIdx.x;
  if (idx >= total) return;
  int d = idx % 10, f = (idx / 10) % 66, b = idx / 660;
  const float* hp = h + (size_t)b * 3300 + f;
  float s = 0.f;
  for (int l = 0; l < 10; ++l) s += T[d * 35 + l] * hp[(size_t)(40 + l) * 66];
  float tl = 0.f;
  for (int l = 10; l < 35; ++l) tl += T[d * 35 + l];
  s += tl * hp[(size_t)49 * 66];
  gcnin[((size_t)b * 66 + f) * 32 + d] = s;
}

// ---------------- dct_att: gcnin cols 10..19 (+ zero pad cols 20..31) ----------------
__global__ void dct_att_kernel(const float* __restrict__ attW, const float* __restrict__ V,
                               float* __restrict__ gcnin, int total) {
  int idx = blockIdx.x * blockDim.x + threadIdx.x;
  if (idx >= total) return;
  int d = idx % 10, f = (idx / 10) % 66, b = idx / 660;
  const float* aw = attW + (size_t)b * 16;
  const float* vp = V + (size_t)b * 16 * 660 + f * 10 + d;
  float s = 0.f;
  for (int k = 0; k < 16; ++k) s += aw[k] * vp[(size_t)k * 660];
  size_t row = (size_t)b * 66 + f;
  gcnin[row * 32 + 10 + d] = s;
  gcnin[row * 32 + 20 + d] = 0.f;
  if (d < 2) gcnin[row * 32 + 30 + d] = 0.f;
}

// ---------------- final IDCT: out[b,l,f] = sum_d dct[d,l] * DCT[b*66+f, d] ----------------
__global__ void idct_kernel(const float* __restrict__ DCT, float* __restrict__ out, int total) {
  __shared__ float T[350];
  for (int i = threadIdx.x; i < 350; i += blockDim.x) T[i] = dct_w(i / 35, i % 35);
  __syncthreads();
  int idx = blockIdx.x * blockDim.x + threadIdx.x;
  if (idx >= total) return;
  int f = idx % 66, l = (idx / 66) % 35, b = idx / 2310;
  const float* dp = DCT + ((size_t)b * 66 + f) * 20;
  float s = 0.f;
  for (int d = 0; d < 10; ++d) s += T[d * 35 + l] * dp[d];
  out[idx] = s;
}

// =====================================================================================
extern "C" void kernel_launch(void* const* d_in, const int* in_sizes, int n_in,
                              void* d_out, int out_size, void* d_ws, size_t ws_size,
                              hipStream_t stream) {
  (void)in_sizes; (void)n_in; (void)out_size; (void)ws_size;
  const float* src    = (const float*)d_in[0];
  const float* Wih    = (const float*)d_in[1];
  const float* bih    = (const float*)d_in[3];
  const float* bhh    = (const float*)d_in[4];
  const float* q_w1   = (const float*)d_in[5];
  const float* q_w2   = (const float*)d_in[6];
  const float* k_w1   = (const float*)d_in[7];
  const float* k_w2   = (const float*)d_in[8];
  const float* gc1_w  = (const float*)d_in[9];
  const float* gc1_at = (const float*)d_in[10];
  const float* gc1_b  = (const float*)d_in[11];
  const float* bn1_g  = (const float*)d_in[12];
  const float* bn1_b  = (const float*)d_in[13];
  const float* blk_w  = (const float*)d_in[14];
  const float* blk_at = (const float*)d_in[15];
  const float* blk_b  = (const float*)d_in[16];
  const float* bbn_g  = (const float*)d_in[17];
  const float* bbn_b  = (const float*)d_in[18];
  const float* gc7_w  = (const float*)d_in[19];
  const float* gc7_at = (const float*)d_in[20];
  const float* gc7_b  = (const float*)d_in[21];

  float* outp = (float*)d_out;            // (1024,35,1,66)
  float* hOut = outp + 2365440;           // (67584,50)
  float* cOut = hOut + 3379200;

  // ---------- workspace layout ----------
  char* ws = (char*)d_ws;
  size_t off = 0;
  auto alloc = [&](size_t bytes) { size_t o = off; off = (off + bytes + 255) & ~(size_t)255; return o; };
  // persistent
  half_t* Wk1  = (half_t*)(ws + alloc(416u * 512 * 2));
  half_t* Wq1  = (half_t*)(ws + alloc(416u * 512 * 2));
  half_t* Wk2  = (half_t*)(ws + alloc(2560u * 512 * 2));
  half_t* Wq2  = (half_t*)(ws + alloc(2560u * 512 * 2));
  half_t* Wg1  = (half_t*)(ws + alloc(32u * 512 * 2));
  half_t* Wblk = (half_t*)(ws + alloc(4u * 512 * 512 * 2));
  half_t* Wg7  = (half_t*)(ws + alloc(512u * 64 * 2));
  half_t* AttP = (half_t*)(ws + alloc(6u * 80 * 96 * 2));
  float*  KeyT = (float*)(ws + alloc((size_t)16384 * 512 * 4));
  float*  Qv   = (float*)(ws + alloc((size_t)1024 * 512 * 4));
  float*  AttW = (float*)(ws + alloc((size_t)1024 * 16 * 4));
  float*  GCNIN= (float*)(ws + alloc((size_t)67584 * 32 * 4));
  float*  DCTO = (float*)(ws + alloc((size_t)67584 * 20 * 4));
  float*  Y    = (float*)(ws + alloc((size_t)67584 * 512 * 4));
  size_t sbase = off;
  // conv-phase scratch (dead before GC phase; aliases GC scratch)
  size_t so = sbase;
  auto salloc = [&](size_t bytes) { size_t o = so; so = (so + bytes + 255) & ~(size_t)255; return o; };
  half_t* A1k = (half_t*)(ws + salloc((size_t)20480 * 416 * 2));
  float*  O1k = (float*)(ws + salloc((size_t)20480 * 512 * 4));
  half_t* A2k = (half_t*)(ws + salloc((size_t)16384 * 2560 * 2));
  half_t* A1q = (half_t*)(ws + salloc((size_t)5120 * 416 * 2));
  float*  O1q = (float*)(ws + salloc((size_t)5120 * 512 * 4));
  half_t* A2q = (half_t*)(ws + salloc((size_t)1024 * 2560 * 2));
  // GC-phase scratch (re-uses same region)
  so = sbase;
  half_t* Y16 = (half_t*)(ws + salloc((size_t)67584 * 512 * 2));
  float*  XW  = (float*)(ws + salloc((size_t)67584 * 512 * 4));
  float*  Y2  = (float*)(ws + salloc((size_t)67584 * 512 * 4));
  float*  Vb  = (float*)(ws + salloc((size_t)1024 * 16 * 660 * 4)); // value DCT (lives between phases)

  const int TPB = 256;
  // ---------- weight packs ----------
  pack_w_kernel<<<(416 * 512) / TPB, TPB, 0, stream>>>(k_w1, Wk1, 512, 396, 416, 512, 1, 416 * 512);
  pack_w_kernel<<<(416 * 512) / TPB, TPB, 0, stream>>>(q_w1, Wq1, 512, 396, 416, 512, 1, 416 * 512);
  pack_w_kernel<<<(2560 * 512) / TPB, TPB, 0, stream>>>(k_w2, Wk2, 512, 2560, 2560, 512, 1, 2560 * 512);
  pack_w_kernel<<<(2560 * 512) / TPB, TPB, 0, stream>>>(q_w2, Wq2, 512, 2560, 2560, 512, 1, 2560 * 512);
  pack_w_kernel<<<(32 * 512) / TPB, TPB, 0, stream>>>(gc1_w, Wg1, 20, 512, 32, 512, 0, 32 * 512);
  for (int u = 0; u < 4; ++u)
    pack_w_kernel<<<(512 * 512) / TPB, TPB, 0, stream>>>(blk_w + (size_t)u * 262144,
        Wblk + (size_t)u * 262144, 512, 512, 512, 512, 0, 512 * 512);
  pack_w_kernel<<<(512 * 64) / TPB, TPB, 0, stream>>>(gc7_w, Wg7, 512, 20, 512, 64, 0, 512 * 64);
  pack_w_kernel<<<30, TPB, 0, stream>>>(gc1_at, AttP + 0 * 7680, 66, 66, 80, 96, 0, 80 * 96);
  for (int u = 0; u < 4; ++u)
    pack_w_kernel<<<30, TPB, 0, stream>>>(blk_at + (size_t)u * 4356, AttP + (size_t)(1 + u) * 7680,
        66, 66, 80, 96, 0, 80 * 96);
  pack_w_kernel<<<30, TPB, 0, stream>>>(gc7_at, AttP + 5 * 7680, 66, 66, 80, 96, 0, 80 * 96);

  // ---------- LSTM ----------
  lstm_kernel<<<(67584 * 50) / TPB, TPB, 0, stream>>>(src, Wih, bih, bhh, hOut, cOut, 67584 * 50);

  // ---------- key / query conv stacks as WMMA GEMMs ----------
  pack_conv1_kernel<<<(20480 * 416) / TPB, TPB, 0, stream>>>(hOut, A1k, 20, 0, 20480 * 416);
  gemm_wmma_kernel<<<dim3(8, 320), 128, 0, stream>>>(A1k, Wk1, O1k, 20480, 512, 416, 1);
  pack_conv2_kernel<<<(16384 * 2560) / TPB, TPB, 0, stream>>>(O1k, A2k, 20, 16, 16384 * 2560);
  gemm_wmma_kernel<<<dim3(8, 256), 128, 0, stream>>>(A2k, Wk2, KeyT, 16384, 512, 2560, 1);

  pack_conv1_kernel<<<(5120 * 416) / TPB, TPB, 0, stream>>>(hOut, A1q, 5, 40, 5120 * 416);
  gemm_wmma_kernel<<<dim3(8, 80), 128, 0, stream>>>(A1q, Wq1, O1q, 5120, 512, 416, 1);
  pack_conv2_kernel<<<(1024 * 2560) / TPB, TPB, 0, stream>>>(O1q, A2q, 5, 1, 1024 * 2560);
  gemm_wmma_kernel<<<dim3(8, 16), 128, 0, stream>>>(A2q, Wq2, Qv, 1024, 512, 2560, 1);

  // ---------- attention over DCT'd value windows ----------
  value_dct_kernel<<<(1024 * 16 * 660) / TPB, TPB, 0, stream>>>(hOut, Vb, 1024 * 16 * 660);
  score_kernel<<<1024, 512, 0, stream>>>(Qv, KeyT, AttW);
  dct_in_kernel<<<(1024 * 660) / TPB, TPB, 0, stream>>>(hOut, GCNIN, 1024 * 660);
  dct_att_kernel<<<(1024 * 660) / TPB, TPB, 0, stream>>>(AttW, Vb, GCNIN, 1024 * 660);

  // ---------- GCN tower ----------
  // gc1
  cvt_f32_f16_kernel<<<(67584 * 32) / TPB, TPB, 0, stream>>>(GCNIN, Y16, 67584 * 32);
  gemm_wmma_kernel<<<dim3(8, 1056), 128, 0, stream>>>(Y16, Wg1, XW, 67584, 512, 32, 0);
  att_wmma_kernel<<<dim3(8, 1024), 128, 0, stream>>>(AttP + 0 * 7680, XW, 512, gc1_b,
      bn1_g, bn1_b, (const float*)nullptr, 0, Y, 512, 512, 0);
  // stages
  for (int s = 0; s < 2; ++s) {
    int u0 = s * 2 + 0, u1 = s * 2 + 1;
    cvt_f32_f16_kernel<<<(67584 * 512) / TPB, TPB, 0, stream>>>(Y, Y16, 67584 * 512);
    gemm_wmma_kernel<<<dim3(8, 1056), 128, 0, stream>>>(Y16, Wblk + (size_t)u0 * 262144, XW, 67584, 512, 512, 0);
    att_wmma_kernel<<<dim3(8, 1024), 128, 0, stream>>>(AttP + (size_t)(1 + u0) * 7680, XW, 512,
        blk_b + (size_t)u0 * 512, bbn_g + (size_t)u0 * 33792, bbn_b + (size_t)u0 * 33792,
        (const float*)nullptr, 0, Y2, 512, 512, 0);
    cvt_f32_f16_kernel<<<(67584 * 512) / TPB, TPB, 0, stream>>>(Y2, Y16, 67584 * 512);
    gemm_wmma_kernel<<<dim3(8, 1056), 128, 0, stream>>>(Y16, Wblk + (size_t)u1 * 262144, XW, 67584, 512, 512, 0);
    att_wmma_kernel<<<dim3(8, 1024), 128, 0, stream>>>(AttP + (size_t)(1 + u1) * 7680, XW, 512,
        blk_b + (size_t)u1 * 512, bbn_g + (size_t)u1 * 33792, bbn_b + (size_t)u1 * 33792,
        Y, 512, Y, 512, 512, 1);   // residual add of y0, in-place per-element
  }
  // gc7 + residual gcn_in
  cvt_f32_f16_kernel<<<(67584 * 512) / TPB, TPB, 0, stream>>>(Y, Y16, 67584 * 512);
  gemm_wmma_kernel<<<dim3(1, 1056), 128, 0, stream>>>(Y16, Wg7, XW, 67584, 64, 512, 0);
  att_wmma_kernel<<<dim3(1, 1024), 128, 0, stream>>>(AttP + 5 * 7680, XW, 64, gc7_b,
      (const float*)nullptr, (const float*)nullptr, GCNIN, 32, DCTO, 20, 20, 2);

  // ---------- final IDCT ----------
  idct_kernel<<<(1024 * 2310) / TPB, TPB, 0, stream>>>(DCTO, outp, 1024 * 2310);
}
